// PronunciationModel_47510928229128
// MI455X (gfx1250) — matrix operand order repaired and checked by reading, not compiled, not run on hardware
//
#include <hip/hip_runtime.h>
#include <math.h>

// ---------------------------------------------------------------------------
// MI455X / gfx1250 implementation. All large GEMMs (SSL linear, conv-as-GEMM,
// QKV, attention, out-proj, FFN, scorer) run on v_wmma_f32_16x16x32_f16 with
// f32 accumulation. Attention is flash-style (no T*T materialization).
// GEMM: 32x64 output tile per wave, register double-buffered k-loop so the
// global_load_b128 clauses for step k+32 overlap the 8 WMMAs of step k.
// ---------------------------------------------------------------------------

typedef _Float16 v16h __attribute__((ext_vector_type(16)));
typedef _Float16 v8h  __attribute__((ext_vector_type(8)));
typedef float    v8f  __attribute__((ext_vector_type(8)));

#define DEVINL static __device__ __forceinline__

constexpr int Bc  = 16;
constexpr int Tc  = 1024;
constexpr int FINc = 1024;
constexpr int SSLc = 512;
constexpr int Dc  = 256;
constexpr int Hc  = 4;
constexpr int HDc = 64;
constexpr int Lc  = 4;
constexpr int Pc  = 128;
constexpr int FFc = 2048;
constexpr int BTc = Bc * Tc;

DEVINL v16h make16(v8h lo, v8h hi) {
  v16h r;
#pragma unroll
  for (int i = 0; i < 8; ++i) { r[i] = lo[i]; r[i + 8] = hi[i]; }
  return r;
}

DEVINL v16h zero16() {
  v16h r;
#pragma unroll
  for (int i = 0; i < 16; ++i) r[i] = (_Float16)0.0f;
  return r;
}

// ---------------------------------------------------------------------------
// Generic WMMA GEMM:  C[M,N] = act( optC + A[M,K] @ W[N,K]^T + bias )
//   - A, W are f16; accumulate f32; output f32 (Cf) or f16 (Ch).
//   - `shift` shifts A rows in time (for conv taps); rows whose t+shift
//     falls outside [0,T) contribute zero (batch-boundary correct).
// One wave computes a 32x64 output tile (2 M-subtiles x 4 N-tiles) with
// double-buffered fragment loads for load/WMMA overlap.
// ---------------------------------------------------------------------------
__global__ __launch_bounds__(256) void gemm_wmma_kernel(
    const _Float16* __restrict__ A, int lda,
    const _Float16* __restrict__ W,            // (N, K) row-major
    const float* __restrict__ bias,            // length N or null
    float* __restrict__ Cf, _Float16* __restrict__ Ch,
    int M, int N, int K, int shift, int Tdim, int relu, int accum) {
  int wid = blockIdx.x * 8 + (threadIdx.x >> 5);
  int tilesN = N >> 6;
  int total = (M >> 5) * tilesN;
  if (wid >= total) return;
  int tn4 = wid % tilesN;
  int tm  = wid / tilesN;
  int lane = threadIdx.x & 31;
  int lrow = lane & 15;
  int g    = lane >> 4;

  v8f acc[2][4];
#pragma unroll
  for (int m = 0; m < 2; ++m)
#pragma unroll
    for (int j = 0; j < 4; ++j)
#pragma unroll
      for (int v = 0; v < 8; ++v) acc[m][j][v] = 0.0f;

  const _Float16* Arow[2];
  bool aok[2];
#pragma unroll
  for (int m = 0; m < 2; ++m) {
    int r = tm * 32 + m * 16 + lrow;
    long long ar = r;
    bool ok = true;
    if (shift != 0) {
      int t  = r % Tdim;
      int ts = t + shift;
      ok = (ts >= 0) && (ts < Tdim);
      ar = (long long)r + shift;
    }
    aok[m]  = ok;
    Arow[m] = A + (size_t)ar * (size_t)lda + g * 8;   // A frag: chunks +0,+16
  }
  const _Float16* Wrow[4];
#pragma unroll
  for (int j = 0; j < 4; ++j)
    Wrow[j] = W + (size_t)(tn4 * 64 + j * 16 + lrow) * (size_t)K + g * 8;

  auto loadA = [&](int k, v16h* a) {
#pragma unroll
    for (int m = 0; m < 2; ++m) {
      if (aok[m])
        a[m] = make16(*(const v8h*)(Arow[m] + k),
                      *(const v8h*)(Arow[m] + k + 16));
      else
        a[m] = zero16();
    }
  };
  auto loadB = [&](int k, v16h* bv) {
#pragma unroll
    for (int j = 0; j < 4; ++j)
      bv[j] = make16(*(const v8h*)(Wrow[j] + k),
                     *(const v8h*)(Wrow[j] + k + 16));
  };
  auto compute = [&](const v16h* a, const v16h* bv) {
#pragma unroll
    for (int m = 0; m < 2; ++m)
#pragma unroll
      for (int j = 0; j < 4; ++j)
        acc[m][j] = __builtin_amdgcn_wmma_f32_16x16x32_f16(
            false, a[m], false, bv[j], (short)0, acc[m][j], false, false);
  };

  v16h A0[2], A1[2], B0[4], B1[4];
  loadA(0, A0);
  loadB(0, B0);
  int k = 0;
  while (true) {
    int k2 = k + 32;
    if (k2 < K) { loadA(k2, A1); loadB(k2, B1); }   // prefetch next step
    compute(A0, B0);
    k = k2;
    if (k >= K) break;
    k2 = k + 32;
    if (k2 < K) { loadA(k2, A0); loadB(k2, B0); }
    compute(A1, B1);
    k = k2;
    if (k >= K) break;
  }

#pragma unroll
  for (int m = 0; m < 2; ++m)
#pragma unroll
    for (int j = 0; j < 4; ++j) {
      int n = tn4 * 64 + j * 16 + lrow;
      float bv = bias ? bias[n] : 0.0f;
#pragma unroll
      for (int v = 0; v < 8; ++v) {
        size_t idx = (size_t)(tm * 32 + m * 16 + g * 8 + v) * (size_t)N + n;
        float val = acc[m][j][v] + bv;
        if (accum) val += Cf[idx];
        if (relu)  val = fmaxf(val, 0.0f);
        if (Ch) Ch[idx] = (_Float16)val;
        else    Cf[idx] = val;
      }
    }
}

// ---------------------------------------------------------------------------
// Flash attention: one wave per (b, h, 16-query tile). Streams over 32-key
// blocks: S = Q K^T via WMMA, online softmax (row reductions via shfl_xor
// within 16-lane groups), P re-striped through 1KB LDS into A-layout, V tile
// cooperatively staged into LDS with coalesced b128 loads, then O += P V via
// WMMA. Output written f16 (consumed by out-proj GEMM).
// ---------------------------------------------------------------------------
__global__ __launch_bounds__(32) void attn_kernel(
    const _Float16* __restrict__ qkv,   // (B*T, 3*D) f16, rows = [q | k | v]
    const int* __restrict__ lens,
    _Float16* __restrict__ outh) {      // (B*T, D) f16
  constexpr int QT = Tc / 16;
  int wid = blockIdx.x;
  int qt = wid % QT;
  int t1 = wid / QT;
  int h  = t1 % Hc;
  int b  = t1 / Hc;
  int lane = threadIdx.x;
  int lrow = lane & 15;
  int g    = lane >> 4;
  int len  = lens[b];
  const float scale = 0.125f;              // 1/sqrt(64)
  const int rowstr = 3 * Dc;

  const _Float16* base = qkv + (size_t)(b * Tc) * rowstr;

  // Q fragments (A operand), hd=64 -> two K=32 fragments
  int q = qt * 16 + lrow;
  v16h qa[2];
#pragma unroll
  for (int kh = 0; kh < 2; ++kh) {
    const _Float16* p = base + (size_t)q * rowstr + h * HDc + kh * 32 + g * 8;
    qa[kh] = make16(*(const v8h*)p, *(const v8h*)(p + 16));
  }

  v8f acc[4];
#pragma unroll
  for (int j = 0; j < 4; ++j)
#pragma unroll
    for (int v = 0; v < 8; ++v) acc[j][v] = 0.0f;

  float mrun[8], lrun[8];
#pragma unroll
  for (int v = 0; v < 8; ++v) { mrun[v] = -INFINITY; lrun[v] = 0.0f; }

  __shared__ _Float16 smem[16][32];
  __shared__ alignas(16) _Float16 vtile[32][64];

  for (int kk = 0; kk < Tc; kk += 32) {
    // protect vtile/smem against the previous iteration's readers
    __syncthreads();
    // --- cooperative stage of V tile (32 keys x 32 head cols) into LDS ---
    {
      const _Float16* vp = base + (size_t)(kk + lane) * rowstr + 2 * Dc + h * HDc;
#pragma unroll
      for (int c = 0; c < 4; ++c)
        *(v8h*)&vtile[lane][c * 8] = *(const v8h*)(vp + c * 8);
    }
    float s0[8], s1[8];
    // --- S tiles (16x16 each) for keys [kk, kk+16) and [kk+16, kk+32) ---
#pragma unroll
    for (int sub = 0; sub < 2; ++sub) {
      int key = kk + sub * 16 + lrow;   // this lane's key column
      v8f c;
#pragma unroll
      for (int v = 0; v < 8; ++v) c[v] = 0.0f;
#pragma unroll
      for (int kh = 0; kh < 2; ++kh) {
        const _Float16* kp =
            base + (size_t)key * rowstr + Dc + h * HDc + kh * 32 + g * 8;
        v16h bf = make16(*(const v8h*)kp, *(const v8h*)(kp + 16));
        c = __builtin_amdgcn_wmma_f32_16x16x32_f16(
            false, qa[kh], false, bf, (short)0, c, false, false);
      }
      bool ok = key < len;
#pragma unroll
      for (int v = 0; v < 8; ++v) {
        float sv = ok ? c[v] * scale : -1e30f;
        if (sub == 0) s0[v] = sv; else s1[v] = sv;
      }
    }
    // --- online softmax (row = g*8+v, spread across 16 lanes of group g) ---
#pragma unroll
    for (int v = 0; v < 8; ++v) {
      float rmax = fmaxf(s0[v], s1[v]);
      for (int off = 1; off < 16; off <<= 1)
        rmax = fmaxf(rmax, __shfl_xor(rmax, off, 32));
      float mnew  = fmaxf(mrun[v], rmax);
      float alpha = expf(mrun[v] - mnew);
      float e0 = expf(s0[v] - mnew);
      float e1 = expf(s1[v] - mnew);
      float ps = e0 + e1;
      for (int off = 1; off < 16; off <<= 1) ps += __shfl_xor(ps, off, 32);
      lrun[v] = lrun[v] * alpha + ps;
      mrun[v] = mnew;
#pragma unroll
      for (int j = 0; j < 4; ++j) acc[j][v] *= alpha;
      s0[v] = e0; s1[v] = e1;
    }
    // --- re-stripe P (C layout -> A layout) through LDS ---
#pragma unroll
    for (int v = 0; v < 8; ++v) {
      smem[g * 8 + v][lrow]      = (_Float16)s0[v];
      smem[g * 8 + v][16 + lrow] = (_Float16)s1[v];
    }
    __syncthreads();
    v16h pa;
#pragma unroll
    for (int e = 0; e < 16; ++e) {
      int kc = g * 16 + e;
      pa[e] = smem[lrow][kc];
    }
    // --- O += P @ V (V tile 32x64 -> 4 B fragments, read from LDS) ---
#pragma unroll
    for (int j = 0; j < 4; ++j) {
      v16h vb;
#pragma unroll
      for (int e = 0; e < 16; ++e)
        vb[e] = vtile[g * 16 + e][j * 16 + lrow];
      acc[j] = __builtin_amdgcn_wmma_f32_16x16x32_f16(
          false, pa, false, vb, (short)0, acc[j], false, false);
    }
  }

#pragma unroll
  for (int v = 0; v < 8; ++v) {
    float inv = 1.0f / lrun[v];
    int row = qt * 16 + g * 8 + v;
#pragma unroll
    for (int j = 0; j < 4; ++j)
      outh[(size_t)(b * Tc + row) * Dc + h * HDc + j * 16 + lrow] =
          (_Float16)(acc[j][v] * inv);
  }
}

// ---------------------------------------------------------------------------
// LayerNorm over last dim N (block = N threads, one row per block).
// Optional residual add; writes f32 (in-place safe) and/or f16.
// ---------------------------------------------------------------------------
__global__ void ln_kernel(const float* __restrict__ X,
                          const float* __restrict__ R,
                          const float* __restrict__ w,
                          const float* __restrict__ bb,
                          float* __restrict__ outF,
                          _Float16* __restrict__ outH, int N) {
  extern __shared__ float red[];
  int row = blockIdx.x, tid = threadIdx.x;
  size_t idx = (size_t)row * N + tid;
  float v = X[idx];
  if (R) v += R[idx];
  red[tid] = v;
  __syncthreads();
  for (int s = N >> 1; s > 0; s >>= 1) {
    if (tid < s) red[tid] += red[tid + s];
    __syncthreads();
  }
  float mean = red[0] * (1.0f / N);
  __syncthreads();
  float d = v - mean;
  red[tid] = d * d;
  __syncthreads();
  for (int s = N >> 1; s > 0; s >>= 1) {
    if (tid < s) red[tid] += red[tid + s];
    __syncthreads();
  }
  float var = red[0] * (1.0f / N);
  float y = d / sqrtf(var + 1e-5f) * w[tid] + bb[tid];
  if (outF) outF[idx] = y;
  if (outH) outH[idx] = (_Float16)y;
}

__global__ void cvt_f32_f16(const float* __restrict__ in,
                            _Float16* __restrict__ out, size_t n) {
  size_t i = (size_t)blockIdx.x * blockDim.x + threadIdx.x;
  size_t stride = (size_t)gridDim.x * blockDim.x;
  for (; i < n; i += stride) out[i] = (_Float16)in[i];
}

// conv_W (D, SSL, 3) -> three contiguous f16 (D, SSL) tap matrices
__global__ void conv_w_split_kernel(const float* __restrict__ W,
                                    _Float16* __restrict__ w0,
                                    _Float16* __restrict__ w1,
                                    _Float16* __restrict__ w2, int n) {
  int i = blockIdx.x * blockDim.x + threadIdx.x;
  if (i < n) {
    w0[i] = (_Float16)W[3 * i + 0];
    w1[i] = (_Float16)W[3 * i + 1];
    w2[i] = (_Float16)W[3 * i + 2];
  }
}

// frame score: per row dot(fs_h[row,:256], fs2_W) + b, masked by length
__global__ __launch_bounds__(256) void frame_score_kernel(
    const _Float16* __restrict__ fsh, const float* __restrict__ w2,
    const float* __restrict__ b2, const int* __restrict__ lens,
    float* __restrict__ out) {
  int row = blockIdx.x * 8 + (threadIdx.x >> 5);
  if (row >= BTc) return;
  int lane = threadIdx.x & 31;
  float s = 0.0f;
#pragma unroll
  for (int i = lane; i < Dc; i += 32)
    s += (float)fsh[(size_t)row * Dc + i] * w2[i];
  for (int off = 1; off < 32; off <<= 1) s += __shfl_xor(s, off, 32);
  if (lane == 0) {
    int b = row >> 10, t = row & (Tc - 1);
    out[row] = (t < lens[b]) ? (s + b2[0]) : 0.0f;
  }
}

// per-batch segment means + utterance/fluency/prosody (tiny scalar work)
__global__ void score_kernel(const float* __restrict__ frame,
                             const int* __restrict__ plen,
                             float* __restrict__ ph, float* __restrict__ utt,
                             float* __restrict__ flu, float* __restrict__ pro) {
  __shared__ float cs[Tc + 1];
  int b = blockIdx.x;
  if (threadIdx.x != 0) return;
  cs[0] = 0.0f;
  for (int t = 0; t < Tc; ++t) cs[t + 1] = cs[t] + frame[b * Tc + t];
  int valid = 1, cum = 0;
  for (int p = 0; p < Pc; ++p) {
    int lp = plen[b * Pc + p];
    if (lp <= 0) valid = 0;
    int eff = valid ? lp : 0;
    int prev = cum;
    cum += eff;
    int e  = cum  < 0 ? 0 : (cum  > Tc ? Tc : cum);
    int st = prev < 0 ? 0 : (prev > Tc ? Tc : prev);
    int cnt = e - st;
    ph[b * Pc + p] = (cnt > 0) ? ((cs[e] - cs[st]) / (float)cnt) : 0.0f;
  }
  float s = 0.0f; int c = 0;
  for (int p = 0; p < Pc; ++p)
    if (plen[b * Pc + p] > 0) { s += ph[b * Pc + p]; ++c; }
  int pc = c > 0 ? c : 1;
  float u = s / (float)pc;
  float vs = 0.0f;
  for (int p = 0; p < Pc; ++p)
    if (plen[b * Pc + p] > 0) { float d = ph[b * Pc + p] - u; vs += d * d; }
  float sd = sqrtf(vs / (float)pc);
  float f = fmaxf(100.0f - fminf(sd * 10.0f, 100.0f), 0.0f);
  utt[b] = u; flu[b] = f; pro[b] = 0.5f * (u + f);
}

// ---------------------------------------------------------------------------
// Orchestration
// ---------------------------------------------------------------------------
static inline void launch_gemm(const _Float16* A, int lda, const _Float16* W,
                               const float* bias, float* Cf, _Float16* Ch,
                               int M, int N, int K, int shift, int relu,
                               int accum, hipStream_t stream) {
  int waves = (M / 32) * (N / 64);
  int blocks = (waves + 7) / 8;
  gemm_wmma_kernel<<<blocks, 256, 0, stream>>>(A, lda, W, bias, Cf, Ch, M, N,
                                               K, shift, Tc, relu, accum);
}

extern "C" void kernel_launch(void* const* d_in, const int* in_sizes, int n_in,
                              void* d_out, int out_size, void* d_ws,
                              size_t ws_size, hipStream_t stream) {
  (void)in_sizes; (void)n_in; (void)out_size; (void)ws_size;

  const float* features = (const float*)d_in[0];
  const int*   feat_len = (const int*)d_in[1];
  const int*   phon_len = (const int*)d_in[2];
  const float* ssl_W  = (const float*)d_in[3];
  const float* ssl_b  = (const float*)d_in[4];
  const float* ln0_w  = (const float*)d_in[5];
  const float* ln0_b  = (const float*)d_in[6];
  const float* conv_W = (const float*)d_in[7];
  const float* conv_b = (const float*)d_in[8];
  const float* qkv_W  = (const float*)d_in[9];
  const float* qkv_b  = (const float*)d_in[10];
  const float* out_W  = (const float*)d_in[11];
  const float* out_b  = (const float*)d_in[12];
  const float* ln1_w  = (const float*)d_in[13];
  const float* ln1_b  = (const float*)d_in[14];
  const float* ff1_W  = (const float*)d_in[15];
  const float* ff1_b  = (const float*)d_in[16];
  const float* ff2_W  = (const float*)d_in[17];
  const float* ff2_b  = (const float*)d_in[18];
  const float* ln2_w  = (const float*)d_in[19];
  const float* ln2_b  = (const float*)d_in[20];
  const float* fs1_W  = (const float*)d_in[21];
  const float* fs1_b  = (const float*)d_in[22];
  const float* fs2_W  = (const float*)d_in[23];
  const float* fs2_b  = (const float*)d_in[24];

  float* out = (float*)d_out;  // [frame(BT) | ph(B*P) | utt(B) | flu(B) | pro(B)]

  char* ws = (char*)d_ws;
  size_t off = 0;
  auto alloc = [&](size_t bytes) -> char* {
    char* p = ws + off;
    off = (off + bytes + 255) & ~(size_t)255;
    return p;
  };

  _Float16* sslW_h = (_Float16*)alloc((size_t)SSLc * FINc * 2);
  _Float16* cw0    = (_Float16*)alloc((size_t)Dc * SSLc * 2);
  _Float16* cw1    = (_Float16*)alloc((size_t)Dc * SSLc * 2);
  _Float16* cw2    = (_Float16*)alloc((size_t)Dc * SSLc * 2);
  _Float16* qkvW_h = (_Float16*)alloc((size_t)Lc * 3 * Dc * Dc * 2);
  _Float16* outW_h = (_Float16*)alloc((size_t)Lc * Dc * Dc * 2);
  _Float16* ff1W_h = (_Float16*)alloc((size_t)Lc * FFc * Dc * 2);
  _Float16* ff2W_h = (_Float16*)alloc((size_t)Lc * Dc * FFc * 2);
  _Float16* fs1W_h = (_Float16*)alloc((size_t)Dc * Dc * 2);
  _Float16* feat_h = (_Float16*)alloc((size_t)BTc * FINc * 2);
  float*    xssl_f = (float*)alloc((size_t)BTc * SSLc * 4);
  _Float16* xssl_h = (_Float16*)alloc((size_t)BTc * SSLc * 2);
  float*    x_f    = (float*)alloc((size_t)BTc * Dc * 4);
  _Float16* x_h    = (_Float16*)alloc((size_t)BTc * Dc * 2);
  _Float16* qkv_h  = (_Float16*)alloc((size_t)BTc * 3 * Dc * 2);
  _Float16* attn_h = (_Float16*)alloc((size_t)BTc * Dc * 2);
  float*    proj_f = (float*)alloc((size_t)BTc * Dc * 4);
  _Float16* ff_h   = (_Float16*)alloc((size_t)BTc * FFc * 2);
  _Float16* fs_h   = (_Float16*)alloc((size_t)BTc * Dc * 2);

  // ---- weight / input conversion to f16 ----
  cvt_f32_f16<<<4096, 256, 0, stream>>>(features, feat_h, (size_t)BTc * FINc);
  cvt_f32_f16<<<512, 256, 0, stream>>>(ssl_W, sslW_h, (size_t)SSLc * FINc);
  conv_w_split_kernel<<<(Dc * SSLc + 255) / 256, 256, 0, stream>>>(
      conv_W, cw0, cw1, cw2, Dc * SSLc);
  cvt_f32_f16<<<1024, 256, 0, stream>>>(qkv_W, qkvW_h, (size_t)Lc * 3 * Dc * Dc);
  cvt_f32_f16<<<512, 256, 0, stream>>>(out_W, outW_h, (size_t)Lc * Dc * Dc);
  cvt_f32_f16<<<2048, 256, 0, stream>>>(ff1_W, ff1W_h, (size_t)Lc * FFc * Dc);
  cvt_f32_f16<<<2048, 256, 0, stream>>>(ff2_W, ff2W_h, (size_t)Lc * Dc * FFc);
  cvt_f32_f16<<<256, 256, 0, stream>>>(fs1_W, fs1W_h, (size_t)Dc * Dc);

  // ---- SSL linear + ReLU, then LN0 (writes f16 for the conv GEMMs) ----
  launch_gemm(feat_h, FINc, sslW_h, ssl_b, xssl_f, nullptr,
              BTc, SSLc, FINc, 0, 1, 0, stream);
  ln_kernel<<<BTc, SSLc, SSLc * sizeof(float), stream>>>(
      xssl_f, nullptr, ln0_w, ln0_b, nullptr, xssl_h, SSLc);

  // ---- conv k=3 as 3 shifted GEMM passes; ReLU on final pass ----
  launch_gemm(xssl_h, SSLc, cw1, conv_b, x_f, nullptr, BTc, Dc, SSLc, 0, 0, 0, stream);
  launch_gemm(xssl_h, SSLc, cw0, nullptr, x_f, nullptr, BTc, Dc, SSLc, -1, 0, 1, stream);
  launch_gemm(xssl_h, SSLc, cw2, nullptr, x_f, nullptr, BTc, Dc, SSLc, +1, 1, 1, stream);
  cvt_f32_f16<<<2048, 256, 0, stream>>>(x_f, x_h, (size_t)BTc * Dc);

  // ---- transformer layers ----
  for (int l = 0; l < Lc; ++l) {
    launch_gemm(x_h, Dc, qkvW_h + (size_t)l * 3 * Dc * Dc, qkv_b + l * 3 * Dc,
                nullptr, qkv_h, BTc, 3 * Dc, Dc, 0, 0, 0, stream);
    attn_kernel<<<Bc * Hc * (Tc / 16), 32, 0, stream>>>(qkv_h, feat_len, attn_h);
    launch_gemm(attn_h, Dc, outW_h + (size_t)l * Dc * Dc, out_b + l * Dc,
                proj_f, nullptr, BTc, Dc, Dc, 0, 0, 0, stream);
    ln_kernel<<<BTc, Dc, Dc * sizeof(float), stream>>>(
        x_f, proj_f, ln1_w + l * Dc, ln1_b + l * Dc, x_f, x_h, Dc);
    launch_gemm(x_h, Dc, ff1W_h + (size_t)l * FFc * Dc, ff1_b + l * FFc,
                nullptr, ff_h, BTc, FFc, Dc, 0, 1, 0, stream);
    launch_gemm(ff_h, FFc, ff2W_h + (size_t)l * Dc * FFc, ff2_b + l * Dc,
                proj_f, nullptr, BTc, Dc, FFc, 0, 0, 0, stream);
    ln_kernel<<<BTc, Dc, Dc * sizeof(float), stream>>>(
        x_f, proj_f, ln2_w + l * Dc, ln2_b + l * Dc, x_f, x_h, Dc);
  }

  // ---- scorer head ----
  launch_gemm(x_h, Dc, fs1W_h, fs1_b, nullptr, fs_h, BTc, Dc, Dc, 0, 1, 0, stream);
  frame_score_kernel<<<BTc / 8, 256, 0, stream>>>(fs_h, fs2_W, fs2_b, feat_len, out);
  score_kernel<<<Bc, 32, 0, stream>>>(out, phon_len,
                                      out + BTc,
                                      out + BTc + Bc * Pc,
                                      out + BTc + Bc * Pc + Bc,
                                      out + BTc + Bc * Pc + 2 * Bc);
}
